// DisentangledSelfAttention_32444182954230
// MI455X (gfx1250) — matrix operand order repaired and checked by reading, not compiled, not run on hardware
//
#include <hip/hip_runtime.h>
#include <hip/hip_bf16.h>
#include <stdint.h>

// ---------------------------------------------------------------------------
// DeBERTa disentangled self-attention for MI455X (gfx1250, wave32, WMMA).
//   K1 x5 : projections  C = A @ W^T + b   (bf16 out, head-major / V-transposed)
//   K2    : fused flash attention with windowed c2p/p2c position GEMMs,
//           K/V tiles staged to LDS via GLOBAL_LOAD_ASYNC_TO_LDS_B128
//   K3    : output projection (f32 out)
// ---------------------------------------------------------------------------

#define NB 4
#define NH 16
#define TT 1024      // T == 2*ATT_SPAN == 1024
#define DK 64
#define TQ 128       // q rows per attention workgroup (8 waves x 16 rows)

#if defined(__has_builtin)
#if __has_builtin(__builtin_amdgcn_global_load_async_to_lds_b128)
#define HAVE_ASYNC 1
#endif
#endif
#ifndef HAVE_ASYNC
#define HAVE_ASYNC 0
#endif

typedef __attribute__((ext_vector_type(16))) __bf16        v16bf;
typedef __attribute__((ext_vector_type(8)))  float         v8f;

union FragU { v16bf bf; uint4 q[2]; };

static __device__ inline v8f vzero() {
  v8f z = {0.f,0.f,0.f,0.f,0.f,0.f,0.f,0.f};
  return z;
}

static __device__ inline unsigned short f2bf(float f) {
  union { float f; unsigned u; } x; x.f = f;
  unsigned r = x.u + 0x7FFFu + ((x.u >> 16) & 1u);   // round-to-nearest-even
  return (unsigned short)(r >> 16);
}

static __device__ inline v8f wmma_bf(v16bf a, v16bf b, v8f c) {
  // (neg_a, A, neg_b, B, c_mod, C, reuse_a, reuse_b)
  return __builtin_amdgcn_wmma_f32_16x16x32_bf16(false, a, false, b, (short)0, c,
                                                 false, false);
}

#if HAVE_ASYNC
// Builtin signature (from clang diagnostic): first arg is
// 'vector_size(16) int __device__ *'  i.e. v4i in addrspace(1); LDS side is v4i
// in addrspace(3).
typedef int v4i __attribute__((vector_size(16)));
static __device__ inline __attribute__((address_space(1))) v4i* as1(const void* p) {
  return (__attribute__((address_space(1))) v4i*)(uintptr_t)p;
}
static __device__ inline __attribute__((address_space(3))) v4i* as3(const void* p) {
  // generic LDS pointer -> 32-bit LDS offset (aperture maps addr[31:0])
  return (__attribute__((address_space(3))) v4i*)(unsigned)(uintptr_t)p;
}
static __device__ inline void wait_async0() {
#if __has_builtin(__builtin_amdgcn_s_wait_asynccnt)
  __builtin_amdgcn_s_wait_asynccnt(0);
#else
  asm volatile("s_wait_asynccnt 0x0" ::: "memory");
#endif
}
#endif

// A-fragment (16x32 bf16, rows = lane%16): element e -> K = kofs + e + (e>=8?8:0) + 8*half
// => two contiguous 8-element runs at kofs+8*half and kofs+16+8*half.
static __device__ inline v16bf ldA(const unsigned short* base, int stride,
                                   int ln, int half, int kofs) {
  const unsigned short* p = base + (size_t)ln * stride + kofs + half * 8;
  FragU u;
  u.q[0] = *reinterpret_cast<const uint4*>(p);
  u.q[1] = *reinterpret_cast<const uint4*>(p + 16);
  return u.bf;
}

// B-fragment (32x16 bf16, col = lane%16): element e -> K = kofs + e + (e>=8?8:0) + 8*half
// => two contiguous 8-element runs per lane; rowptr is the per-lane row base.
static __device__ inline v16bf ldB(const unsigned short* rowptr, int half, int kofs) {
  const unsigned short* p = rowptr + kofs + half * 8;
  FragU u;
  u.q[0] = *reinterpret_cast<const uint4*>(p);
  u.q[1] = *reinterpret_cast<const uint4*>(p + 16);
  return u.bf;
}

// ---------------------------------------------------------------------------
// GEMM: C(MxN) = A(Mx1024) @ W(Nx1024)^T + bias,  N = 1024, K = 1024.
// mode 0: bf16 out, head-major   idx = ((b*16+h)*1024 + t)*64 + d
// mode 1: bf16 out, V-transposed idx = ((b*16+h)*64 + d)*1024 + t
// mode 2: f32  out, row-major    idx = m*1024 + n
// ---------------------------------------------------------------------------
__global__ __launch_bounds__(256) void gemm_nt(const float* __restrict__ A,
                                               const float* __restrict__ W,
                                               const float* __restrict__ bias,
                                               float* __restrict__ outF,
                                               unsigned short* __restrict__ outB,
                                               int mode) {
  __shared__ alignas(16) unsigned short As[128 * 40];   // 128 rows x 32 K, stride 40
  __shared__ alignas(16) unsigned short Bs[128 * 40];

  const int tid  = threadIdx.x;
  const int m0   = blockIdx.y * 128, n0 = blockIdx.x * 128;
  const int w    = tid >> 5, l = tid & 31, ln = l & 15, half = l >> 4;
  const int wm   = w & 3, wn = w >> 2;          // 4 (M) x 2 (N) waves
  const int lrow = tid >> 1, lcol = (tid & 1) * 16;

  v8f acc[2][4];
#pragma unroll
  for (int i = 0; i < 2; i++)
#pragma unroll
    for (int j = 0; j < 4; j++) acc[i][j] = vzero();

  const float* Ap = A + (size_t)(m0 + lrow) * 1024 + lcol;
  const float* Wp = W + (size_t)(n0 + lrow) * 1024 + lcol;
  unsigned* asd = (unsigned*)&As[lrow * 40 + lcol];
  unsigned* bsd = (unsigned*)&Bs[lrow * 40 + lcol];

  for (int kk = 0; kk < 1024; kk += 32) {
#pragma unroll
    for (int i = 0; i < 4; i++) {
      float4 av = *reinterpret_cast<const float4*>(Ap + kk + i * 4);
      float4 bv = *reinterpret_cast<const float4*>(Wp + kk + i * 4);
      asd[2*i]   = (unsigned)f2bf(av.x) | ((unsigned)f2bf(av.y) << 16);
      asd[2*i+1] = (unsigned)f2bf(av.z) | ((unsigned)f2bf(av.w) << 16);
      bsd[2*i]   = (unsigned)f2bf(bv.x) | ((unsigned)f2bf(bv.y) << 16);
      bsd[2*i+1] = (unsigned)f2bf(bv.z) | ((unsigned)f2bf(bv.w) << 16);
    }
    __syncthreads();
    v16bf af[2], bfr[4];
#pragma unroll
    for (int mi = 0; mi < 2; mi++)
      af[mi] = ldA(As + (size_t)(wm * 32 + mi * 16) * 40, 40, ln, half, 0);
#pragma unroll
    for (int nf = 0; nf < 4; nf++)
      bfr[nf] = ldB(Bs + (size_t)(wn * 64 + nf * 16 + ln) * 40, half, 0);
#pragma unroll
    for (int mi = 0; mi < 2; mi++)
#pragma unroll
      for (int nf = 0; nf < 4; nf++)
        acc[mi][nf] = wmma_bf(af[mi], bfr[nf], acc[mi][nf]);
    __syncthreads();
  }

#pragma unroll
  for (int mi = 0; mi < 2; mi++)
#pragma unroll
    for (int nf = 0; nf < 4; nf++) {
      const int n = n0 + wn * 64 + nf * 16 + ln;
      const float bn = bias[n];
#pragma unroll
      for (int r = 0; r < 8; r++) {
        const int m = m0 + wm * 32 + mi * 16 + r + half * 8;
        const float v = acc[mi][nf][r] + bn;
        if (mode == 2) {
          outF[(size_t)m * 1024 + n] = v;
        } else {
          const int bb = m >> 10, t = m & 1023;
          const int h = n >> 6, d = n & 63;
          size_t idx;
          if (mode == 0) idx = (((size_t)(bb * 16 + h)) * 1024 + t) * 64 + d;
          else           idx = (((size_t)(bb * 16 + h)) * 64 + d) * 1024 + t;
          outB[idx] = f2bf(v);
        }
      }
    }
}

// ---------------------------------------------------------------------------
// Fused flash attention with disentangled relative-position terms.
// Grid (T/128, H, B), 256 threads = 8 waves, 16 q-rows per wave.
// ---------------------------------------------------------------------------
__global__ __launch_bounds__(256) void attn_disent(
    const unsigned short* __restrict__ Qh, const unsigned short* __restrict__ Kh,
    const unsigned short* __restrict__ Vt, const unsigned short* __restrict__ Pk,
    const unsigned short* __restrict__ Pq, float* __restrict__ Ctx) {
  __shared__ alignas(16) float          sY[8][16 * 33];   // per-wave gather scratch
  __shared__ alignas(16) unsigned short sP[8][16 * 72];   // per-wave 16x64 bf16 probs
#if HAVE_ASYNC
  __shared__ alignas(16) unsigned short sK[64 * DK];      // [k_local][d]  8 KB
  __shared__ alignas(16) unsigned short sV[DK * 64];      // [d][t_local]  8 KB
#endif

  const int tid = threadIdx.x;
  const int w = tid >> 5, l = tid & 31, ln = l & 15, half = l >> 4;
  const int qb = blockIdx.x * TQ;
  const int h = blockIdx.y, b = blockIdx.z;
  const int bh = b * NH + h;

  const unsigned short* Qb  = Qh + (size_t)bh * TT * DK;
  const unsigned short* Kb  = Kh + (size_t)bh * TT * DK;
  const unsigned short* Vb  = Vt + (size_t)bh * DK * TT;   // [d][t]
  const unsigned short* Pkh = Pk + (size_t)h * TT * DK;    // [s][d]
  const unsigned short* Pqh = Pq + (size_t)h * TT * DK;

  float*          sYw = sY[w];
  unsigned short* sPw = sP[w];
  const int qw = qb + w * 16;            // wave's first global q row

  v16bf aq[2];
#pragma unroll
  for (int c = 0; c < 2; c++) aq[c] = ldA(Qb + (size_t)qw * DK, DK, ln, half, c * 32);

  float rowM[8], rowS[8];
#pragma unroll
  for (int r = 0; r < 8; r++) { rowM[r] = -1e30f; rowS[r] = 0.f; }
  v8f o[4];
#pragma unroll
  for (int df = 0; df < 4; df++) o[df] = vzero();

  const float scale = 0.07216878364870323f;   // 1/sqrt(64*3)

  for (int k0 = 0; k0 < TT; k0 += 64) {
#if HAVE_ASYNC
    {
      // K tile: rows k0..k0+63 are one contiguous 8 KB region.
      const char* gk = (const char*)(Kb + (size_t)k0 * DK);
      const int off = tid * 16;
      __builtin_amdgcn_global_load_async_to_lds_b128(as1(gk + off),
                                                     as3((char*)sK + off), 0, 0);
      __builtin_amdgcn_global_load_async_to_lds_b128(as1(gk + off + 4096),
                                                     as3((char*)sK + off + 4096), 0, 0);
      // V tile: 64 d-rows x 128 B from the transposed V layout.
#pragma unroll
      for (int i = 0; i < 2; i++) {
        const int c = tid + 256 * i;
        const int d = c >> 3, sub = (c & 7) * 16;
        __builtin_amdgcn_global_load_async_to_lds_b128(
            as1((const char*)(Vb + (size_t)d * TT + k0) + sub),
            as3((char*)sV + d * 128 + sub), 0, 0);
      }
      wait_async0();
    }
    __syncthreads();
#endif

    v8f S[4];
#pragma unroll
    for (int nf = 0; nf < 4; nf++) S[nf] = vzero();

    // ---- content-to-content: S += Q @ K^T --------------------------------
#pragma unroll
    for (int nf = 0; nf < 4; nf++) {
#if HAVE_ASYNC
      const unsigned short* krow = sK + (size_t)(nf * 16 + ln) * DK;
#else
      const unsigned short* krow = Kb + (size_t)(k0 + nf * 16 + ln) * DK;
#endif
#pragma unroll
      for (int c = 0; c < 2; c++)
        S[nf] = wmma_bf(aq[c], ldB(krow, half, c * 32), S[nf]);
    }

    // ---- c2p / p2c via 32-wide windowed position GEMMs + LDS gather ------
    for (int nf = 0; nf < 4; nf++) {
      const int win = k0 + nf * 16 - qw + 512 - 16;  // s = win + j, j = 16+kc-qc
      int s0 = win + ln;      s0 = s0 < 0 ? 0 : (s0 > 1023 ? 1023 : s0);
      int s1 = win + 16 + ln; s1 = s1 < 0 ? 0 : (s1 > 1023 ? 1023 : s1);

      // c2p: Y[qc][j] = Q[qw+qc] . Pk[clamp(win+j)]
      v8f y0 = vzero(), y1 = vzero();
#pragma unroll
      for (int c = 0; c < 2; c++) {
        y0 = wmma_bf(aq[c], ldB(Pkh + (size_t)s0 * DK, half, c * 32), y0);
        y1 = wmma_bf(aq[c], ldB(Pkh + (size_t)s1 * DK, half, c * 32), y1);
      }
#pragma unroll
      for (int r = 0; r < 8; r++) {
        sYw[(r + half * 8) * 33 + ln]      = y0[r];
        sYw[(r + half * 8) * 33 + 16 + ln] = y1[r];
      }
      __asm volatile("" ::: "memory");
#pragma unroll
      for (int r = 0; r < 8; r++) {
        const int qc = r + half * 8;
        S[nf][r] += sYw[qc * 33 + (16 + ln - qc)];
      }
      __asm volatile("" ::: "memory");

      // p2c: Z[kc][j] = K[k0+16nf+kc] . Pq[clamp(win+j)]
      v8f z0 = vzero(), z1 = vzero();
#pragma unroll
      for (int c = 0; c < 2; c++) {
#if HAVE_ASYNC
        v16bf akc = ldA(sK + (size_t)(nf * 16) * DK, DK, ln, half, c * 32);
#else
        v16bf akc = ldA(Kb + (size_t)(k0 + nf * 16) * DK, DK, ln, half, c * 32);
#endif
        z0 = wmma_bf(akc, ldB(Pqh + (size_t)s0 * DK, half, c * 32), z0);
        z1 = wmma_bf(akc, ldB(Pqh + (size_t)s1 * DK, half, c * 32), z1);
      }
#pragma unroll
      for (int r = 0; r < 8; r++) {
        sYw[(r + half * 8) * 33 + ln]      = z0[r];
        sYw[(r + half * 8) * 33 + 16 + ln] = z1[r];
      }
      __asm volatile("" ::: "memory");
#pragma unroll
      for (int r = 0; r < 8; r++) {
        const int qc = r + half * 8;
        S[nf][r] += sYw[ln * 33 + (16 + ln - qc)];
      }
      __asm volatile("" ::: "memory");
    }

    // ---- scale + online softmax (per-row over 64 keys) -------------------
    float nm[8], al[8], ps[8];
#pragma unroll
    for (int r = 0; r < 8; r++) {
#pragma unroll
      for (int nf = 0; nf < 4; nf++) S[nf][r] *= scale;
      nm[r] = fmaxf(fmaxf(S[0][r], S[1][r]), fmaxf(S[2][r], S[3][r]));
    }
#pragma unroll
    for (int r = 0; r < 8; r++)
      for (int msk = 1; msk < 16; msk <<= 1)
        nm[r] = fmaxf(nm[r], __shfl_xor(nm[r], msk, 32));
#pragma unroll
    for (int r = 0; r < 8; r++) {
      const float mn = fmaxf(rowM[r], nm[r]);
      al[r] = __expf(rowM[r] - mn);
      rowM[r] = mn;
      float s = 0.f;
#pragma unroll
      for (int nf = 0; nf < 4; nf++) {
        const float p = __expf(S[nf][r] - mn);
        S[nf][r] = p;
        s += p;
      }
      ps[r] = s;
    }
#pragma unroll
    for (int r = 0; r < 8; r++)
      for (int msk = 1; msk < 16; msk <<= 1)
        ps[r] += __shfl_xor(ps[r], msk, 32);
#pragma unroll
    for (int r = 0; r < 8; r++) rowS[r] = rowS[r] * al[r] + ps[r];
#pragma unroll
    for (int df = 0; df < 4; df++)
#pragma unroll
      for (int r = 0; r < 8; r++) o[df][r] *= al[r];

    // ---- P (bf16) via LDS -> A-fragments; O += P @ V ---------------------
#pragma unroll
    for (int nf = 0; nf < 4; nf++)
#pragma unroll
      for (int r = 0; r < 8; r++)
        sPw[(r + half * 8) * 72 + nf * 16 + ln] = f2bf(S[nf][r]);
    __asm volatile("" ::: "memory");
    v16bf ap[2];
#pragma unroll
    for (int c = 0; c < 2; c++) ap[c] = ldA(sPw, 72, ln, half, c * 32);
#pragma unroll
    for (int df = 0; df < 4; df++) {
#if HAVE_ASYNC
      const unsigned short* vrow = sV + (size_t)(df * 16 + ln) * 64;
#else
      const unsigned short* vrow = Vb + (size_t)(df * 16 + ln) * TT + k0;
#endif
#pragma unroll
      for (int c = 0; c < 2; c++)
        o[df] = wmma_bf(ap[c], ldB(vrow, half, c * 32), o[df]);
    }
    __asm volatile("" ::: "memory");
#if HAVE_ASYNC
    __syncthreads();   // protect sK/sV before next tile's async writes
#endif
  }

  // ---- normalize + store ctx[b][q][h*64+d] -------------------------------
#pragma unroll
  for (int df = 0; df < 4; df++)
#pragma unroll
    for (int r = 0; r < 8; r++) {
      const int q = qw + r + half * 8;
      Ctx[((size_t)(b * TT + q)) * 1024 + h * 64 + df * 16 + ln] =
          o[df][r] / rowS[r];
    }
}

// ---------------------------------------------------------------------------
extern "C" void kernel_launch(void* const* d_in, const int* in_sizes, int n_in,
                              void* d_out, int out_size, void* d_ws, size_t ws_size,
                              hipStream_t stream) {
  (void)in_sizes; (void)n_in; (void)out_size; (void)ws_size;
  const float* x   = (const float*)d_in[0];
  const float* Wq  = (const float*)d_in[1];
  const float* bq  = (const float*)d_in[2];
  const float* Wk  = (const float*)d_in[3];
  const float* bk  = (const float*)d_in[4];
  const float* Wv  = (const float*)d_in[5];
  const float* bv  = (const float*)d_in[6];
  const float* Wo  = (const float*)d_in[7];
  const float* bo  = (const float*)d_in[8];
  const float* Wpk = (const float*)d_in[9];
  const float* bpk = (const float*)d_in[10];
  const float* Wpq = (const float*)d_in[11];
  const float* bpq = (const float*)d_in[12];
  const float* rel = (const float*)d_in[13];

  char* ws = (char*)d_ws;                                  // 44 MB used
  unsigned short* Qh = (unsigned short*)(ws + 0);          // 8 MB  bf16 [b][h][t][d]
  unsigned short* Kh = (unsigned short*)(ws + (8u  << 20));// 8 MB
  unsigned short* Vt = (unsigned short*)(ws + (16u << 20));// 8 MB  bf16 [b][h][d][t]
  unsigned short* Pk = (unsigned short*)(ws + (24u << 20));// 2 MB  bf16 [h][s][d]
  unsigned short* Pq = (unsigned short*)(ws + (26u << 20));// 2 MB
  float*          Cx = (float*)(ws + (28u << 20));         // 16 MB f32 [b][t][1024]

  dim3 blk(256);
  gemm_nt<<<dim3(8, 32), blk, 0, stream>>>(x,   Wq,  bq,  nullptr, Qh, 0);
  gemm_nt<<<dim3(8, 32), blk, 0, stream>>>(x,   Wk,  bk,  nullptr, Kh, 0);
  gemm_nt<<<dim3(8, 32), blk, 0, stream>>>(x,   Wv,  bv,  nullptr, Vt, 1);
  gemm_nt<<<dim3(8, 8),  blk, 0, stream>>>(rel, Wpk, bpk, nullptr, Pk, 0);
  gemm_nt<<<dim3(8, 8),  blk, 0, stream>>>(rel, Wpq, bpq, nullptr, Pq, 0);
  attn_disent<<<dim3(TT / TQ, NH, NB), blk, 0, stream>>>(Qh, Kh, Vt, Pk, Pq, Cx);
  gemm_nt<<<dim3(8, 32), blk, 0, stream>>>(Cx,  Wo,  bo,  (float*)d_out, nullptr, 2);
}
